// LatentAttention_78297253806417
// MI455X (gfx1250) — matrix operand order repaired and checked by reading, not compiled, not run on hardware
//
#include <hip/hip_runtime.h>
#include <hip/hip_bf16.h>

// MLA forward for MI455X (gfx1250): bf16 WMMA, f32 accumulate, b128 data
// movement, async global->LDS staging (ASYNCcnt) in the attention kernel.
// d_out = [out: B*S*H f32][latent: B*S*L f32]
// d_ws  = [q bf16][k bf16][vT bf16][wT_dkv][wT_q][wT_k][wT_v]  (~26 MB)

typedef __attribute__((ext_vector_type(16))) __bf16 v16bf;
typedef __attribute__((ext_vector_type(8)))  __bf16 v8bf;
typedef __attribute__((ext_vector_type(4)))  __bf16 v4bf;
typedef __attribute__((ext_vector_type(8)))  float  v8f;

#define B_  8
#define S_  4096
#define E_  2048
#define L_  64
#define H_  128
#define BS_ (B_ * S_)

__device__ __forceinline__ v8f wmma_bf16(v16bf a, v16bf b, v8f c) {
  return __builtin_amdgcn_wmma_f32_16x16x32_bf16(
      false, a, false, b, (short)0, c, false, false);
}

__device__ __forceinline__ v16bf cat8(v8bf lo, v8bf hi) {
  return __builtin_shufflevector(lo, hi, 0, 1, 2, 3, 4, 5, 6, 7,
                                 8, 9, 10, 11, 12, 13, 14, 15);
}

// CDNA5 async global->LDS copy (16B per lane), tracked by ASYNCcnt.
// vdst = LDS byte offset (flat LDS addresses truncate to the low 32 bits).
__device__ __forceinline__ void async_to_lds_b128(const __bf16* lds_dst,
                                                  const __bf16* gsrc) {
  const uint32_t loff = (uint32_t)(uintptr_t)lds_dst;
  const uint64_t gaddr = (uint64_t)(uintptr_t)gsrc;
  asm volatile("global_load_async_to_lds_b128 %0, %1, off"
               :
               : "v"(loff), "v"(gaddr)
               : "memory");
}
__device__ __forceinline__ void wait_asynccnt0() {
  asm volatile("s_wait_asynccnt 0x0" ::: "memory");
}

// ---------------------------------------------------------------------------
// Kernel 0: one-time weight transpose+downcast: wT[col][k] bf16.
// Makes every WMMA B-fragment a contiguous 32-byte load.
// ---------------------------------------------------------------------------
__global__ __launch_bounds__(256) void prep_weights(
    const float* __restrict__ w_dkv, const float* __restrict__ w_k,
    const float* __restrict__ w_v, const float* __restrict__ w_q,
    __bf16* __restrict__ wT_dkv, __bf16* __restrict__ wT_q,
    __bf16* __restrict__ wT_k, __bf16* __restrict__ wT_v) {
  int i = blockIdx.x * 256 + threadIdx.x;
  if (i < L_ * E_) { wT_dkv[i] = (__bf16)w_dkv[(size_t)(i % E_) * L_ + i / E_]; return; }
  i -= L_ * E_;
  if (i < H_ * E_) { wT_q[i] = (__bf16)w_q[(size_t)(i % E_) * H_ + i / E_]; return; }
  i -= H_ * E_;
  if (i < H_ * L_) { wT_k[i] = (__bf16)w_k[(size_t)(i % L_) * H_ + i / L_]; return; }
  i -= H_ * L_;
  if (i < H_ * L_) { wT_v[i] = (__bf16)w_v[(size_t)(i % L_) * H_ + i / L_]; return; }
}

// ---------------------------------------------------------------------------
// Kernel 1: fused projections, 64-row tile, 384 threads = 12 waves.
//  waves 0..3  -> latent col tiles (64 cols):  latent = x @ w_dkv
//  waves 4..11 -> q col tiles (128 cols):      q      = x @ w_q  (pre-scaled)
// Then k,v = latent @ w_k/w_v from the LDS latent tile.
// x staging is software-pipelined through registers: next chunk's float4
// loads issue before this chunk's WMMAs, cvt+ds_store happen next iteration.
// ---------------------------------------------------------------------------
__global__ __launch_bounds__(384) void mla_proj_kernel(
    const float* __restrict__ x, const __bf16* __restrict__ wT_dkv,
    const __bf16* __restrict__ wT_q, const __bf16* __restrict__ wT_k,
    const __bf16* __restrict__ wT_v, float* __restrict__ lat_out,
    __bf16* __restrict__ q_ws, __bf16* __restrict__ k_ws,
    __bf16* __restrict__ v_wsT) {
  __shared__ __bf16 xs[64 * 32];    // x K-chunk (64 rows x 32 dims)
  __shared__ __bf16 lat_s[64 * 64]; // latent tile

  const int tid   = threadIdx.x;
  const int wave  = tid >> 5;
  const int lane  = tid & 31;
  const int khalf = lane >> 4;
  const int lidx  = lane & 15;
  const int r0    = blockIdx.x * 64;

  const bool is_lat = (wave < 4);
  const __bf16* wT  = is_lat ? wT_dkv : wT_q;
  const int col0    = is_lat ? wave * 16 : (wave - 4) * 16;

  v8f zero = {};
  v8f acc[4];
#pragma unroll
  for (int r = 0; r < 4; ++r) acc[r] = zero;

  // staging assignment: threads 0..255 each own 2 float4 chunks per K-step
  const bool stager = (tid < 256);
  const int e0 = tid * 4, e1 = (tid + 256) * 4;
  const int row0 = e0 >> 5, cx0 = e0 & 31;
  const int row1 = e1 >> 5, cx1 = e1 & 31;
  float4 c0, c1;
  if (stager) {
    c0 = *(const float4*)&x[(size_t)(r0 + row0) * E_ + cx0];
    c1 = *(const float4*)&x[(size_t)(r0 + row1) * E_ + cx1];
  }

  for (int kc = 0; kc < E_ / 32; ++kc) {
    if (stager) {
      v4bf o0, o1;
      o0[0] = (__bf16)c0.x; o0[1] = (__bf16)c0.y;
      o0[2] = (__bf16)c0.z; o0[3] = (__bf16)c0.w;
      o1[0] = (__bf16)c1.x; o1[1] = (__bf16)c1.y;
      o1[2] = (__bf16)c1.z; o1[3] = (__bf16)c1.w;
      *(v4bf*)&xs[row0 * 32 + cx0] = o0;
      *(v4bf*)&xs[row1 * 32 + cx1] = o1;
    }
    __syncthreads();

    if (stager && kc + 1 < E_ / 32) { // issue next chunk early
      c0 = *(const float4*)&x[(size_t)(r0 + row0) * E_ + (kc + 1) * 32 + cx0];
      c1 = *(const float4*)&x[(size_t)(r0 + row1) * E_ + (kc + 1) * 32 + cx1];
    }

    const v16bf bw =
        *(const v16bf*)&wT[(size_t)(col0 + lidx) * E_ + kc * 32 + khalf * 16];
#pragma unroll
    for (int r = 0; r < 4; ++r) {
      const v8bf lo = *(const v8bf*)&xs[(r * 16 + lidx) * 32 + khalf * 16];
      const v8bf hi = *(const v8bf*)&xs[(r * 16 + lidx) * 32 + khalf * 16 + 8];
      acc[r] = wmma_bf16(cat8(lo, hi), bw, acc[r]);
    }
    __syncthreads();
  }

  if (is_lat) {
#pragma unroll
    for (int r = 0; r < 4; ++r)
#pragma unroll
      for (int j = 0; j < 8; ++j) {
        const int row = r * 16 + j + 8 * khalf;
        lat_out[(size_t)(r0 + row) * L_ + col0 + lidx] = acc[r][j];
        lat_s[row * 64 + col0 + lidx] = (__bf16)acc[r][j];
      }
  } else {
    const float scale = 0.08838834764831845f; // 1/sqrt(128), folded into q
#pragma unroll
    for (int r = 0; r < 4; ++r)
#pragma unroll
      for (int j = 0; j < 8; ++j) {
        const int row = r * 16 + j + 8 * khalf;
        q_ws[(size_t)(r0 + row) * H_ + col0 + lidx] = (__bf16)(acc[r][j] * scale);
      }
  }
  __syncthreads();

  // Phase 2: k,v = latent @ w_k / w_v (K=64 -> 2 WMMA steps), waves 0..7.
  if (wave < 8) {
    const int c0w = wave * 16;
    const size_t wrow = (size_t)(c0w + lidx) * L_;
    const v16bf bk0 = *(const v16bf*)&wT_k[wrow + khalf * 16];
    const v16bf bk1 = *(const v16bf*)&wT_k[wrow + 32 + khalf * 16];
    const v16bf bv0 = *(const v16bf*)&wT_v[wrow + khalf * 16];
    const v16bf bv1 = *(const v16bf*)&wT_v[wrow + 32 + khalf * 16];
    const int bbatch = r0 >> 12; // r0 / S_
    const int s0     = r0 & (S_ - 1);
#pragma unroll
    for (int r = 0; r < 4; ++r) {
      v8f ak = zero, av = zero;
#pragma unroll
      for (int kc = 0; kc < 2; ++kc) {
        const v8bf lo =
            *(const v8bf*)&lat_s[(r * 16 + lidx) * 64 + kc * 32 + khalf * 16];
        const v8bf hi =
            *(const v8bf*)&lat_s[(r * 16 + lidx) * 64 + kc * 32 + khalf * 16 + 8];
        const v16bf a = cat8(lo, hi);
        ak = wmma_bf16(a, kc ? bk1 : bk0, ak);
        av = wmma_bf16(a, kc ? bv1 : bv0, av);
      }
#pragma unroll
      for (int j = 0; j < 8; ++j) {
        const int row = r * 16 + j + 8 * khalf;
        k_ws[(size_t)(r0 + row) * H_ + c0w + lidx] = (__bf16)ak[j];
        // V stored dim-major: attention's P@V B-fragment becomes contiguous
        v_wsT[(size_t)bbatch * H_ * S_ + (size_t)(c0w + lidx) * S_ + s0 + row] =
            (__bf16)av[j];
      }
    }
  }
}

// ---------------------------------------------------------------------------
// Kernel 2: causal flash attention. Grid (S/64, B), 128 threads = 4 waves.
// K/V tiles staged with GLOBAL_LOAD_ASYNC_TO_LDS_B128 into double-buffered
// LDS, one tile ahead; single barrier per key tile. Blocks are mapped
// longest-work-first (reversed qx) to balance the causal triangle.
// ---------------------------------------------------------------------------
__global__ __launch_bounds__(128) void mla_attn_kernel(
    const __bf16* __restrict__ q_ws, const __bf16* __restrict__ k_ws,
    const __bf16* __restrict__ v_wsT, float* __restrict__ out) {
  __shared__ __bf16 ks[2][32 * 128];   // [buf][key][dim]
  __shared__ __bf16 vs_t[2][128 * 32]; // [buf][dim][key]
  __shared__ __bf16 ps[4][16 * 32];    // per-wave P scratch

  const int tid   = threadIdx.x;
  const int wave  = tid >> 5;
  const int lane  = tid & 31;
  const int khalf = lane >> 4;
  const int lidx  = lane & 15;
  const int bb    = blockIdx.y;
  const int qx    = gridDim.x - 1 - blockIdx.x; // heaviest tiles launch first
  const int q0    = qx * 64 + wave * 16;
  const size_t base  = (size_t)bb * S_;
  const size_t vbase = (size_t)bb * H_ * S_;

  // Q fragments (pre-scaled by 1/sqrt(d)): contiguous v8bf loads
  v16bf qf[4];
  const size_t qrow = (base + q0 + lidx) * H_;
#pragma unroll
  for (int kc = 0; kc < 4; ++kc) {
    const v8bf lo = *(const v8bf*)&q_ws[qrow + kc * 32 + khalf * 16];
    const v8bf hi = *(const v8bf*)&q_ws[qrow + kc * 32 + khalf * 16 + 8];
    qf[kc] = cat8(lo, hi);
  }

  v8f zero = {};
  v8f acc[8];
  float m[8], lsum[8];
#pragma unroll
  for (int t = 0; t < 8; ++t) acc[t] = zero;
#pragma unroll
  for (int j = 0; j < 8; ++j) { m[j] = -__builtin_inff(); lsum[j] = 0.f; }

  const int nkt = qx * 2 + 2;

  // async stage of one 32-key tile (K key-major, V dim-major), 8x16B per lane
  auto stage_tile = [&](int j0s, int buf) {
#pragma unroll
    for (int rep = 0; rep < 4; ++rep) {
      const int e = (tid + rep * 128) * 8;
      const int r = e >> 7, c = e & 127;
      async_to_lds_b128(&ks[buf][e], &k_ws[(base + j0s + r) * H_ + c]);
    }
#pragma unroll
    for (int rep = 0; rep < 4; ++rep) {
      const int e = (tid + rep * 128) * 8;
      const int d = e >> 5, ko = e & 31;
      async_to_lds_b128(&vs_t[buf][e], &v_wsT[vbase + (size_t)d * S_ + j0s + ko]);
    }
  };

  stage_tile(0, 0); // prologue

  for (int kt = 0; kt < nkt; ++kt) {
    const int j0 = kt * 32;
    const int buf = kt & 1;
    wait_asynccnt0();  // this tile's async writes have landed in LDS
    __syncthreads();   // ...and the other buffer is no longer being read
    if (kt + 1 < nkt) stage_tile(j0 + 32, 1 - buf); // run one tile ahead

    if (j0 <= q0 + 15) { // wave-uniform skip of fully masked tiles
      v8f s0 = zero, s1 = zero;
#pragma unroll
      for (int kc = 0; kc < 4; ++kc) {
        const v16bf b0 =
            *(const v16bf*)&ks[buf][lidx * 128 + kc * 32 + khalf * 16];
        const v16bf b1 =
            *(const v16bf*)&ks[buf][(16 + lidx) * 128 + kc * 32 + khalf * 16];
        s0 = wmma_bf16(qf[kc], b0, s0);
        s1 = wmma_bf16(qf[kc], b1, s1);
      }

      float alpha[8];
#pragma unroll
      for (int j = 0; j < 8; ++j) {
        const int qr = q0 + j + 8 * khalf;
        const bool msk0 = (j0 + lidx) > qr;
        const bool msk1 = (j0 + 16 + lidx) > qr;
        float v0 = msk0 ? -__builtin_inff() : s0[j];
        float v1 = msk1 ? -__builtin_inff() : s1[j];
        float tmax = fmaxf(v0, v1);
        tmax = fmaxf(tmax, __shfl_xor(tmax, 1));
        tmax = fmaxf(tmax, __shfl_xor(tmax, 2));
        tmax = fmaxf(tmax, __shfl_xor(tmax, 4));
        tmax = fmaxf(tmax, __shfl_xor(tmax, 8));
        const float mn = fmaxf(m[j], tmax);
        alpha[j] = (mn == -__builtin_inff()) ? 1.f : __expf(m[j] - mn);
        const float p0 = msk0 ? 0.f : __expf(v0 - mn);
        const float p1 = msk1 ? 0.f : __expf(v1 - mn);
        float rs = p0 + p1;
        rs += __shfl_xor(rs, 1);
        rs += __shfl_xor(rs, 2);
        rs += __shfl_xor(rs, 4);
        rs += __shfl_xor(rs, 8);
        lsum[j] = lsum[j] * alpha[j] + rs;
        m[j] = mn;
        ps[wave][(j + 8 * khalf) * 32 + lidx]      = (__bf16)p0;
        ps[wave][(j + 8 * khalf) * 32 + 16 + lidx] = (__bf16)p1;
      }
#pragma unroll
      for (int t = 0; t < 8; ++t)
#pragma unroll
        for (int j = 0; j < 8; ++j) acc[t][j] *= alpha[j];

      const v8bf plo = *(const v8bf*)&ps[wave][lidx * 32 + khalf * 16];
      const v8bf phi = *(const v8bf*)&ps[wave][lidx * 32 + khalf * 16 + 8];
      const v16bf ap = cat8(plo, phi);
#pragma unroll
      for (int t = 0; t < 8; ++t) {
        const v16bf bv =
            *(const v16bf*)&vs_t[buf][(t * 16 + lidx) * 32 + khalf * 16];
        acc[t] = wmma_bf16(ap, bv, acc[t]);
      }
    }
  }

#pragma unroll
  for (int j = 0; j < 8; ++j) {
    const float inv = 1.f / lsum[j];
    const int row = q0 + j + 8 * khalf;
#pragma unroll
    for (int t = 0; t < 8; ++t)
      out[(base + row) * H_ + t * 16 + lidx] = acc[t][j] * inv;
  }
}

// ---------------------------------------------------------------------------
extern "C" void kernel_launch(void* const* d_in, const int* in_sizes, int n_in,
                              void* d_out, int out_size, void* d_ws,
                              size_t ws_size, hipStream_t stream) {
  const float* x     = (const float*)d_in[0];
  const float* w_dkv = (const float*)d_in[1];
  const float* w_k   = (const float*)d_in[2];
  const float* w_v   = (const float*)d_in[3];
  const float* w_q   = (const float*)d_in[4];

  float* out     = (float*)d_out;
  float* lat_out = out + (size_t)BS_ * H_;

  __bf16* q_ws   = (__bf16*)d_ws;
  __bf16* k_ws   = q_ws + (size_t)BS_ * H_;
  __bf16* v_wsT  = k_ws + (size_t)BS_ * H_;
  __bf16* wT_dkv = v_wsT + (size_t)BS_ * H_;
  __bf16* wT_q   = wT_dkv + (size_t)L_ * E_;
  __bf16* wT_k   = wT_q + (size_t)H_ * E_;
  __bf16* wT_v   = wT_k + (size_t)H_ * L_;

  const int prep_elems = L_ * E_ + H_ * E_ + 2 * H_ * L_; // 409600
  prep_weights<<<dim3((prep_elems + 255) / 256), dim3(256), 0, stream>>>(
      w_dkv, w_k, w_v, w_q, wT_dkv, wT_q, wT_k, wT_v);
  mla_proj_kernel<<<dim3(BS_ / 64), dim3(384), 0, stream>>>(
      x, wT_dkv, wT_q, wT_k, wT_v, lat_out, q_ws, k_ws, v_wsT);
  mla_attn_kernel<<<dim3(S_ / 64, B_), dim3(128), 0, stream>>>(
      q_ws, k_ws, v_wsT, out);
}
